// BiPointNet2SSGCls_55147380081268
// MI455X (gfx1250) — matrix operand order repaired and checked by reading, not compiled, not run on hardware
//
#include <hip/hip_runtime.h>

typedef __attribute__((ext_vector_type(16))) _Float16 v16h;
typedef __attribute__((ext_vector_type(8)))  _Float16 h8;
typedef __attribute__((ext_vector_type(8)))  float    v8f;

#define EPSBN 1e-5f
#define BATCH 16

// ---------------------------------------------------------------------------
// Farthest point sampling: one block per batch, dists cached in LDS,
// block argmax with lowest-index tie-break (jnp.argmax semantics).
// ---------------------------------------------------------------------------
__global__ void fps_kernel(const float* __restrict__ pos, int* __restrict__ cidx,
                           int n, int npoints) {
  const int b = blockIdx.x;
  const float* P = pos + (size_t)b * n * 3;
  int* out = cidx + (size_t)b * npoints;
  __shared__ float dists[4096];
  __shared__ float rd[256];
  __shared__ int   ri[256];
  __shared__ int   s_last;
  const int tid = threadIdx.x;
  for (int i = tid; i < n; i += 256) dists[i] = 1e10f;
  if (tid == 0) { s_last = 0; out[0] = 0; }
  __syncthreads();
  for (int it = 1; it < npoints; ++it) {
    const int last = s_last;
    const float lx = P[last*3+0], ly = P[last*3+1], lz = P[last*3+2];
    float best = -1.0f; int bi = n;
    for (int i = tid; i < n; i += 256) {
      float dx = P[i*3+0]-lx, dy = P[i*3+1]-ly, dz = P[i*3+2]-lz;
      float d = dx*dx + dy*dy + dz*dz;
      float nd = fminf(dists[i], d);
      dists[i] = nd;
      if (nd > best) { best = nd; bi = i; }   // ascending i keeps first max
    }
    rd[tid] = best; ri[tid] = bi;
    __syncthreads();
    for (int s = 128; s > 0; s >>= 1) {
      if (tid < s) {
        if (rd[tid+s] > rd[tid] || (rd[tid+s] == rd[tid] && ri[tid+s] < ri[tid])) {
          rd[tid] = rd[tid+s]; ri[tid] = ri[tid+s];
        }
      }
      __syncthreads();
    }
    if (tid == 0) { s_last = ri[0]; out[it] = ri[0]; }
    __syncthreads();
  }
}

__global__ void gather_pos_kernel(const float* __restrict__ pos, const int* __restrict__ cidx,
                                  float* __restrict__ newpos, int n, int S) {
  int i = blockIdx.x * blockDim.x + threadIdx.x;
  int total = BATCH * S * 3;
  if (i >= total) return;
  int b = i / (S * 3); int r = i % (S * 3); int s = r / 3; int c = r % 3;
  newpos[i] = pos[((size_t)b * n + cidx[b * S + s]) * 3 + c];
}

// Ball query: first-k in index order within radius, pad with first hit.
__global__ void ballq_kernel(const float* __restrict__ cent, const float* __restrict__ pos,
                             int* __restrict__ nidx, int n, int S, int K, float r2) {
  int i = blockIdx.x * blockDim.x + threadIdx.x;
  if (i >= BATCH * S) return;
  int b = i / S;
  const float* P = pos + (size_t)b * n * 3;
  float cx = cent[i*3+0], cy = cent[i*3+1], cz = cent[i*3+2];
  int* out = nidx + (size_t)i * K;
  int cnt = 0;
  for (int j = 0; j < n && cnt < K; ++j) {
    float dx = P[j*3+0]-cx, dy = P[j*3+1]-cy, dz = P[j*3+2]-cz;
    if (dx*dx + dy*dy + dz*dz < r2) out[cnt++] = j;
  }
  int pad = (cnt > 0) ? out[0] : 0;
  for (; cnt < K; ++cnt) out[cnt] = pad;
}

// Build grouped f16 input rows: [relpos(3) | feat(featC) | zero pad -> CinPad]
__global__ void group_kernel(const float* __restrict__ pos, const float* __restrict__ newpos,
                             const float* __restrict__ feat, const int* __restrict__ nidx,
                             _Float16* __restrict__ out, int n, int S, int K,
                             int featC, int CinPad, int total) {
  int row = blockIdx.x * blockDim.x + threadIdx.x;
  if (row >= total) return;
  int b = row / (S * K);
  int s = (row / K) % S;
  int idx = nidx[row];
  const float* pp = pos + ((size_t)b * n + idx) * 3;
  const float* np = newpos + ((size_t)b * S + s) * 3;
  _Float16* o = out + (size_t)row * CinPad;
  o[0] = (_Float16)(pp[0]-np[0]);
  o[1] = (_Float16)(pp[1]-np[1]);
  o[2] = (_Float16)(pp[2]-np[2]);
  int c = 3;
  if (featC > 0) {
    const float* f = feat + ((size_t)b * n + idx) * featC;
    for (int j = 0; j < featC; ++j) o[3 + j] = (_Float16)f[j];
    c = 3 + featC;
  }
  for (; c < CinPad; ++c) o[c] = (_Float16)0.f;
}

// SA3 group_all: rows = B*SP, [pos(3) | feat(featC) | pad]
__global__ void concat_kernel(const float* __restrict__ pos, const float* __restrict__ feat,
                              _Float16* __restrict__ out, int featC, int CinPad, int total) {
  int row = blockIdx.x * blockDim.x + threadIdx.x;
  if (row >= total) return;
  _Float16* o = out + (size_t)row * CinPad;
  o[0] = (_Float16)pos[row*3+0];
  o[1] = (_Float16)pos[row*3+1];
  o[2] = (_Float16)pos[row*3+2];
  for (int j = 0; j < featC; ++j) o[3 + j] = (_Float16)feat[(size_t)row * featC + j];
  for (int c = 3 + featC; c < CinPad; ++c) o[c] = (_Float16)0.f;
}

__global__ void wconv_kernel(const float* __restrict__ W, _Float16* __restrict__ Wh,
                             int Cout, int Cin, int CinPad) {
  int i = blockIdx.x * blockDim.x + threadIdx.x;
  if (i >= Cout * CinPad) return;
  int o = i / CinPad, c = i % CinPad;
  Wh[i] = (c < Cin) ? (_Float16)W[(size_t)o * Cin + c] : (_Float16)0.f;
}

// ---------------------------------------------------------------------------
// WMMA GEMM: Y[M,Cout] = X[M,CinPad] * Wh[Cout,CinPad]^T  (f16 in, f32 acc)
// Each wave owns a 16x16 tile and walks 4 consecutive row tiles; per-column
// sum / sum^2 accumulated for BN stats with one atomic pair per column.
// A (16x32 f16): lane<16 -> halves {K0..7, K16..23}; lane>=16 -> {K8..15, K24..31}
// B (32x16 f16): lane<16 -> {K0..7, K16..23} of col=lane; lane>=16 -> {K8..15, K24..31} of col=lane-16
// D (16x16 f32): vgpr r -> row r (lane<16) / row r+8 (lane>=16), col = lane%16
// ---------------------------------------------------------------------------
__global__ void mm_kernel(const _Float16* __restrict__ X, const _Float16* __restrict__ Wh,
                          _Float16* __restrict__ Y, float* __restrict__ stats,
                          int M, int CinPad, int Cout) {
  const int lane = threadIdx.x & 31;
  const int wave = threadIdx.x >> 5;
  const bool hi = lane >= 16;
  const int ln = hi ? lane - 16 : lane;           // 0..15
  const int colBase = blockIdx.x * 16;
  const int colW = colBase + ln;
  const _Float16* wrow = Wh + (size_t)colW * CinPad;

  float colSum = 0.f, colSq = 0.f;
  for (int t = 0; t < 4; ++t) {
    int rt = (blockIdx.y * 8 + wave) * 4 + t;     // row tile index
    int rowBase = rt * 16;
    if (rowBase >= M) break;                      // uniform across wave
    const _Float16* xrow = X + (size_t)(rowBase + ln) * CinPad;
    v8f acc = {};
    for (int k0 = 0; k0 < CinPad; k0 += 32) {
      const h8* ap = (const h8*)(xrow + k0 + (hi ? 8 : 0));
      h8 a_lo = ap[0];
      h8 a_hi = ap[2];                            // +16 halves
      const h8* bp = (const h8*)(wrow + k0 + (hi ? 8 : 0));
      h8 b_lo = bp[0];
      h8 b_hi = bp[2];
      v16h a = __builtin_shufflevector(a_lo, a_hi, 0,1,2,3,4,5,6,7,8,9,10,11,12,13,14,15);
      v16h bv = __builtin_shufflevector(b_lo, b_hi, 0,1,2,3,4,5,6,7,8,9,10,11,12,13,14,15);
      acc = __builtin_amdgcn_wmma_f32_16x16x32_f16(false, a, false, bv,
                                                   (short)0, acc, false, false);
    }
    for (int r = 0; r < 8; ++r) {
      int row = rowBase + r + (hi ? 8 : 0);
      float v = acc[r];
      Y[(size_t)row * Cout + colW] = (_Float16)v;
      colSum += v; colSq += v * v;
    }
  }
  colSum += __shfl_xor(colSum, 16);
  colSq  += __shfl_xor(colSq, 16);
  if (!hi) {
    atomicAdd(&stats[colW], colSum);
    atomicAdd(&stats[Cout + colW], colSq);
  }
}

__global__ void bn_relu_kernel(_Float16* __restrict__ Y, const float* __restrict__ stats,
                               const float* __restrict__ g, const float* __restrict__ b,
                               int M, int Cout) {
  size_t i = (size_t)blockIdx.x * blockDim.x + threadIdx.x;
  size_t total = (size_t)M * Cout;
  if (i >= total) return;
  int c = (int)(i % Cout);
  float mean = stats[c] / (float)M;
  float var  = stats[Cout + c] / (float)M - mean * mean;
  float inv  = rsqrtf(var + EPSBN);
  float v = (float)Y[i];
  v = g[c] * (v - mean) * inv + b[c];
  Y[i] = (_Float16)(v > 0.f ? v : 0.f);
}

__global__ void bn_relu_maxpool_kernel(const _Float16* __restrict__ Y, const float* __restrict__ stats,
                                       const float* __restrict__ g, const float* __restrict__ b,
                                       float* __restrict__ pooled, int M, int Cout, int Kpool) {
  size_t i = (size_t)blockIdx.x * blockDim.x + threadIdx.x;
  size_t total = (size_t)M * Cout;
  if (i >= total) return;
  int c = (int)(i % Cout);
  int row = (int)(i / Cout);
  float mean = stats[c] / (float)M;
  float var  = stats[Cout + c] / (float)M - mean * mean;
  float inv  = rsqrtf(var + EPSBN);
  float v = (float)Y[i];
  v = g[c] * (v - mean) * inv + b[c];
  v = v > 0.f ? v : 0.f;
  int grp = row / Kpool;
  atomicMax((int*)&pooled[(size_t)grp * Cout + c], __float_as_int(v));  // v>=0 -> bit-monotone
}

// ---------------------------------------------------------------------------
// Binarized head: out[b,o] = (sign(x[b,:]) . sign(W[o,:])) * mean|W[o,:]|
// ---------------------------------------------------------------------------
__global__ void binlin_kernel(const float* __restrict__ X, const float* __restrict__ W,
                              float* __restrict__ out, int Cin, int Cout) {
  int i = blockIdx.x * blockDim.x + threadIdx.x;
  if (i >= BATCH * Cout) return;
  int o = i % Cout, bi = i / Cout;
  const float* wr = W + (size_t)o * Cin;
  const float* xr = X + (size_t)bi * Cin;
  float s = 0.f, asum = 0.f;
  for (int c = 0; c < Cin; ++c) {
    float wv = wr[c], xv = xr[c];
    asum += fabsf(wv);
    float sw = (float)((wv > 0.f) - (wv < 0.f));
    float sx = (float)((xv > 0.f) - (xv < 0.f));
    s += sx * sw;
  }
  out[i] = s * (asum / (float)Cin);
}

__global__ void bn_relu_batch_kernel(float* __restrict__ X, const float* __restrict__ g,
                                     const float* __restrict__ b, int C) {
  int c = blockIdx.x * blockDim.x + threadIdx.x;
  if (c >= C) return;
  float m = 0.f;
  for (int i = 0; i < BATCH; ++i) m += X[(size_t)i * C + c];
  m /= (float)BATCH;
  float v = 0.f;
  for (int i = 0; i < BATCH; ++i) { float d = X[(size_t)i * C + c] - m; v += d * d; }
  v /= (float)BATCH;
  float inv = rsqrtf(v + EPSBN);
  for (int i = 0; i < BATCH; ++i) {
    float val = g[c] * (X[(size_t)i * C + c] - m) * inv + b[c];
    X[(size_t)i * C + c] = val > 0.f ? val : 0.f;
  }
}

// ---------------------------------------------------------------------------
static inline int cdiv(long long a, int b) { return (int)((a + b - 1) / b); }

extern "C" void kernel_launch(void* const* d_in, const int* in_sizes, int n_in,
                              void* d_out, int out_size, void* d_ws, size_t ws_size,
                              hipStream_t stream) {
  (void)in_sizes; (void)n_in; (void)out_size; (void)ws_size;
  auto F = [&](int i) { return (const float*)d_in[i]; };
  const float* x = F(0);
  // sa1: w/g/b * 3 -> 1..9 ; sa2 -> 10..18 ; sa3 -> 19..27 ; head 28..34

  // ---- workspace arena (256B aligned) ----
  size_t off = 0;
  auto alloc = [&](size_t bytes) {
    void* p = (char*)d_ws + off;
    off += (bytes + 255) & ~(size_t)255;
    return p;
  };
  int*      cidx1   = (int*)alloc((size_t)BATCH * 512 * 4);
  float*    pos1    = (float*)alloc((size_t)BATCH * 512 * 3 * 4);
  int*      nidx1   = (int*)alloc((size_t)BATCH * 512 * 64 * 4);
  int*      cidx2   = (int*)alloc((size_t)BATCH * 128 * 4);
  float*    pos2    = (float*)alloc((size_t)BATCH * 128 * 3 * 4);
  int*      nidx2   = (int*)alloc((size_t)BATCH * 128 * 64 * 4);
  _Float16* bufG    = (_Float16*)alloc((size_t)131072 * 160 * 2);   // 42 MB (max grouped)
  _Float16* bufY1   = (_Float16*)alloc((size_t)524288 * 64 * 2);    // 67 MB
  _Float16* bufY2   = (_Float16*)alloc((size_t)524288 * 64 * 2);    // 67 MB
  _Float16* bufY3   = (_Float16*)alloc((size_t)524288 * 128 * 2);   // 134 MB
  float*    pooled1 = (float*)alloc((size_t)BATCH * 512 * 128 * 4);
  float*    pooled2 = (float*)alloc((size_t)BATCH * 128 * 256 * 4);
  float*    pooled3 = (float*)alloc((size_t)BATCH * 1024 * 4);
  float*    stats   = (float*)alloc((size_t)2 * 1024 * 4);
  _Float16* whBuf   = (_Float16*)alloc((size_t)1024 * 512 * 2);
  float*    h1      = (float*)alloc((size_t)BATCH * 512 * 4);
  float*    h2      = (float*)alloc((size_t)BATCH * 256 * 4);

  auto run_layer = [&](const _Float16* Xb, const float* Wf, const float* g, const float* bb,
                       _Float16* Yb, int Mr, int Cin, int CinPad, int Cout,
                       float* pooledOut, int Kpool) {
    wconv_kernel<<<cdiv((long long)Cout * CinPad, 256), 256, 0, stream>>>(Wf, whBuf, Cout, Cin, CinPad);
    hipMemsetAsync(stats, 0, (size_t)2 * Cout * 4, stream);
    dim3 grid(Cout / 16, cdiv(Mr, 512));
    mm_kernel<<<grid, 256, 0, stream>>>(Xb, whBuf, Yb, stats, Mr, CinPad, Cout);
    long long tot = (long long)Mr * Cout;
    if (pooledOut) {
      hipMemsetAsync(pooledOut, 0, (size_t)(Mr / Kpool) * Cout * 4, stream);
      bn_relu_maxpool_kernel<<<cdiv(tot, 256), 256, 0, stream>>>(Yb, stats, g, bb, pooledOut, Mr, Cout, Kpool);
    } else {
      bn_relu_kernel<<<cdiv(tot, 256), 256, 0, stream>>>(Yb, stats, g, bb, Mr, Cout);
    }
  };

  // ================= SA1: N=4096 -> S=512, r=0.2, K=64, MLP 3->64->64->128 ==
  fps_kernel<<<BATCH, 256, 0, stream>>>(x, cidx1, 4096, 512);
  gather_pos_kernel<<<cdiv((long long)BATCH * 512 * 3, 256), 256, 0, stream>>>(x, cidx1, pos1, 4096, 512);
  ballq_kernel<<<cdiv((long long)BATCH * 512, 256), 256, 0, stream>>>(pos1, x, nidx1, 4096, 512, 64, 0.2f * 0.2f);
  {
    int M1 = BATCH * 512 * 64;   // 524288
    group_kernel<<<cdiv(M1, 256), 256, 0, stream>>>(x, pos1, nullptr, nidx1, bufG, 4096, 512, 64, 0, 32, M1);
    run_layer(bufG,  F(1), F(2), F(3), bufY1, M1,  3, 32,  64, nullptr, 0);
    run_layer(bufY1, F(4), F(5), F(6), bufY2, M1, 64, 64,  64, nullptr, 0);
    run_layer(bufY2, F(7), F(8), F(9), bufY3, M1, 64, 64, 128, pooled1, 64);
  }

  // ================= SA2: 512 -> 128, r=0.4, K=64, MLP 131->128->128->256 ===
  fps_kernel<<<BATCH, 256, 0, stream>>>(pos1, cidx2, 512, 128);
  gather_pos_kernel<<<cdiv((long long)BATCH * 128 * 3, 256), 256, 0, stream>>>(pos1, cidx2, pos2, 512, 128);
  ballq_kernel<<<cdiv((long long)BATCH * 128, 256), 256, 0, stream>>>(pos2, pos1, nidx2, 512, 128, 64, 0.4f * 0.4f);
  {
    int M2 = BATCH * 128 * 64;   // 131072
    group_kernel<<<cdiv(M2, 256), 256, 0, stream>>>(pos1, pos2, pooled1, nidx2, bufG, 512, 128, 64, 128, 160, M2);
    run_layer(bufG,  F(10), F(11), F(12), bufY1, M2, 131, 160, 128, nullptr, 0);
    run_layer(bufY1, F(13), F(14), F(15), bufY2, M2, 128, 128, 128, nullptr, 0);
    run_layer(bufY2, F(16), F(17), F(18), bufY3, M2, 128, 128, 256, pooled2, 64);
  }

  // ================= SA3 (group_all): 128 pts, MLP 259->256->512->1024 ======
  {
    int M3 = BATCH * 128;        // 2048
    concat_kernel<<<cdiv(M3, 256), 256, 0, stream>>>(pos2, pooled2, bufG, 256, 288, M3);
    run_layer(bufG,  F(19), F(20), F(21), bufY1, M3, 259, 288,  256, nullptr, 0);
    run_layer(bufY1, F(22), F(23), F(24), bufY2, M3, 256, 256,  512, nullptr, 0);
    run_layer(bufY2, F(25), F(26), F(27), bufY3, M3, 512, 512, 1024, pooled3, 128);
  }

  // ================= Binarized head ========================================
  binlin_kernel<<<cdiv((long long)BATCH * 512, 256), 256, 0, stream>>>(pooled3, F(28), h1, 1024, 512);
  bn_relu_batch_kernel<<<cdiv(512, 256), 256, 0, stream>>>(h1, F(29), F(30), 512);
  binlin_kernel<<<cdiv((long long)BATCH * 256, 256), 256, 0, stream>>>(h1, F(31), h2, 512, 256);
  bn_relu_batch_kernel<<<cdiv(256, 256), 256, 0, stream>>>(h2, F(32), F(33), 256);
  binlin_kernel<<<cdiv((long long)BATCH * 40, 256), 256, 0, stream>>>(h2, F(34), (float*)d_out, 256, 40);
}
